// LeapfrogIntegrator_74440373174356
// MI455X (gfx1250) — matrix operand order repaired and checked by reading, not compiled, not run on hardware
//
#include <hip/hip_runtime.h>

// CDNA5 / gfx1250 leapfrog integrator with low-rank Christoffel via WMMA.
// One wave (32 lanes) owns a 16-row batch strip; all 8 steps run on-chip.

typedef __attribute__((ext_vector_type(16))) _Float16 v16h;
typedef __attribute__((ext_vector_type(8)))  float    v8f;
typedef float f32x4 __attribute__((ext_vector_type(4), may_alias));

#define DFEAT  256   // feature dim D
#define RRANK  32    // low rank R
#define NWAVES 4     // waves per block (1 per SIMD32 of the WGP)
#define VSTR   264   // sV row stride (halfs): 528B = 16B-aligned, 4 banks/row offset
#define USTR   264   // sU row stride (halfs)
#define TSTR   40    // sT row stride (halfs): 80B = 16B-aligned, 20 banks/row offset

__device__ __forceinline__ v16h ld_pair16(const _Float16* p0, const _Float16* p1) {
  union { v16h h; f32x4 q[2]; } u;
  u.q[0] = *(const f32x4*)p0;   // 16B chunk -> frag elements 0..7 (VGPR 0..3)
  u.q[1] = *(const f32x4*)p1;   // 16B chunk -> frag elements 8..15 (VGPR 4..7)
  return u.h;
}

__device__ __forceinline__ v8f wmma_f16(v16h a, v16h b, v8f c) {
  // D(16x16,f32) = A(16x32,f16) * B(32x16,f16) + C
  return __builtin_amdgcn_wmma_f32_16x16x32_f16(false, a, false, b, (short)0, c,
                                                false, false);
}

__global__ __launch_bounds__(128, 1)
void leapfrog_wmma_kernel(const float* __restrict__ xin,
                          const float* __restrict__ vin,
                          const float* __restrict__ fin,
                          const float* __restrict__ Uin,   // (D,R) row-major
                          const float* __restrict__ Win,   // (R,D) row-major
                          const int*   __restrict__ stepsPtr,
                          float* __restrict__ xout,
                          float* __restrict__ vout)
{
  __shared__ __align__(16) _Float16 sU[RRANK * USTR];        // U^T: [r][d]
  __shared__ __align__(16) _Float16 sV[NWAVES * 16 * VSTR];  // v (f16), per-wave strip
  __shared__ __align__(16) _Float16 sT[NWAVES * 16 * TSTR];  // t^2 staging, per-wave

  const int tid  = threadIdx.x;
  const int wave = tid >> 5;
  const int lane = tid & 31;
  const int ll   = lane & 15;   // lane % 16
  const int lh   = lane >> 4;   // 0 = low half, 1 = high half

  _Float16* sVw = sV + wave * (16 * VSTR);
  _Float16* sTw = sT + wave * (16 * TSTR);

  const int rowbase = (blockIdx.x * NWAVES + wave) * 16;

  // ---- one-time staging: U^T into persistent LDS ----
  for (int i = tid; i < DFEAT * RRANK; i += 128) {
    int d = i >> 5, r = i & 31;               // U[d][r]
    sU[r * USTR + d] = (_Float16)Uin[i];
  }
  // ---- one-time staging: W^T into sV scratch (overwritten later by v) ----
  for (int i = tid; i < RRANK * DFEAT; i += 128) {
    int r = i >> 8, c = i & 255;              // W[r][c]
    sV[c * RRANK + r] = (_Float16)Win[i];
  }
  __syncthreads();

  // ---- W B-fragments resident in registers (16 frags x 8 VGPRs) ----
  // B 32x16 layout: lane -> col c = jn*16+ll ; K rows contiguous, hi lanes +16.
  v16h wf[16];
  #pragma unroll
  for (int jn = 0; jn < 16; ++jn) {
    const _Float16* p = sV + (jn * 16 + ll) * RRANK + lh * 16;
    wf[jn] = ld_pair16(p, p + 8);             // 32 contiguous bytes
  }
  __syncthreads();                            // done with W scratch

  // ---- load state in C-layout: element (row g+8*lh, col j*16+ll) ----
  v8f vc[16], fc[16], xa[16];
  #pragma unroll
  for (int j = 0; j < 16; ++j) {
    #pragma unroll
    for (int g = 0; g < 8; ++g) {
      int lr = g + 8 * lh;
      size_t idx = (size_t)(rowbase + lr) * DFEAT + j * 16 + ll;
      float vv = vin[idx];
      vc[j][g] = vv;
      fc[j][g] = fin[idx];
      xa[j][g] = 0.0f;
      sVw[lr * VSTR + j * 16 + ll] = (_Float16)vv;   // f16 copy for A-frags
    }
  }

  const float h = 0.5f * 0.01f;   // 0.5 * DT * DT_SCALE
  const int nsteps = *stepsPtr;

  for (int s = 0; s < nsteps; ++s) {
    #pragma unroll
    for (int phase = 0; phase < 2; ++phase) {
      // ---------- GEMM1: t(16x32) = v(16x256) @ U(256x32), f32 accum ----------
      v8f t0 = {}, t1 = {};
      #pragma unroll
      for (int k = 0; k < 8; ++k) {
        // A 16x32 f16 frag: lane row = ll; lo lanes K {0..15}, hi {16..31}
        const _Float16* pa = sVw + ll * VSTR + k * 32 + 16 * lh;
        v16h a = ld_pair16(pa, pa + 8);
        // B frags from U^T rows (contiguous K per lane)
        const _Float16* pb0 = sU + ll * USTR        + k * 32 + 16 * lh;
        const _Float16* pb1 = sU + (16 + ll) * USTR + k * 32 + 16 * lh;
        t0 = wmma_f16(a, ld_pair16(pb0, pb0 + 8), t0);
        t1 = wmma_f16(a, ld_pair16(pb1, pb1 + 8), t1);
      }

      // ---------- square in f32, stage t^2 as f16 A-matrix (row-major) ----------
      #pragma unroll
      for (int g = 0; g < 8; ++g) {
        int row = g + 8 * lh;
        float q0 = t0[g] * t0[g];
        float q1 = t1[g] * t1[g];
        sTw[row * TSTR + ll]      = (_Float16)q0;
        sTw[row * TSTR + 16 + ll] = (_Float16)q1;
      }

      // ---------- GEMM2: gamma(16x256) = t2(16x32) @ W(32x256) + update ----------
      const _Float16* pt = sTw + ll * TSTR + 16 * lh;
      v16h a2 = ld_pair16(pt, pt + 8);        // same frag for every N-tile
      #pragma unroll
      for (int jn = 0; jn < 16; ++jn) {
        v8f cz = {};
        v8f gamma = wmma_f16(a2, wf[jn], cz); // one WMMA covers all of K=32
        #pragma unroll
        for (int g = 0; g < 8; ++g) {
          float vh = vc[jn][g] + h * (fc[jn][g] - gamma[g]);  // mu = 0
          vc[jn][g] = vh;
          if (phase == 0) xa[jn][g] += vh;    // x += dt * v_half (deferred)
          sVw[(g + 8 * lh) * VSTR + jn * 16 + ll] = (_Float16)vh;
        }
      }
      // per-wave LDS only; wave-order LDS => no barrier needed
    }
  }

  // ---- finalize: x_out = x_in + dt * sum(v_half); v_out = v ----
  #pragma unroll
  for (int j = 0; j < 16; ++j) {
    #pragma unroll
    for (int g = 0; g < 8; ++g) {
      int lr = g + 8 * lh;
      size_t idx = (size_t)(rowbase + lr) * DFEAT + j * 16 + ll;
      xout[idx] = xin[idx] + 0.01f * xa[j][g];
      vout[idx] = vc[j][g];
    }
  }
}

extern "C" void kernel_launch(void* const* d_in, const int* in_sizes, int n_in,
                              void* d_out, int out_size, void* d_ws, size_t ws_size,
                              hipStream_t stream) {
  const float* x = (const float*)d_in[0];
  const float* v = (const float*)d_in[1];
  const float* f = (const float*)d_in[2];
  const float* U = (const float*)d_in[3];
  const float* W = (const float*)d_in[4];
  const int* steps = (const int*)d_in[5];

  const int B = in_sizes[0] / DFEAT;          // 32768
  float* xo = (float*)d_out;
  float* vo = xo + (size_t)B * DFEAT;

  const int rows_per_block = NWAVES * 16;     // 64
  const int grid = B / rows_per_block;        // 512

  leapfrog_wmma_kernel<<<grid, 128, 0, stream>>>(x, v, f, U, W, steps, xo, vo);
}
